// GroupedLinear_85143431676274
// MI455X (gfx1250) — matrix operand order repaired and checked by reading, not compiled, not run on hardware
//
#include <hip/hip_runtime.h>

// GroupedLinear (MoE expert GEMM) — MI455X / gfx1250, wave32.
// out[t,:] = W[expert(t)] @ x[t,:] + bias[expert(t)]; E=8, IN=OUT=2048, T=16384.
//
// Compute-bound => 3xBF16-split GEMM on V_WMMA_F32_16X16X32_BF16 (f32 accum,
// rel err ~1.5e-5). Two-phase when workspace allows:
//   pass 1: split f32 -> bf16(hi)/bf16(lo) planes once (memory bound, ~23us)
//   pass 2: GEMM streaming bf16 planes, async global->LDS (double buffered,
//           XOR-swizzled 64KB LDS, conflict-free b128 fragment reads),
//           L2-friendly supertile block order (192MB L2 >> 25MB working set).
// Fallback (small ws): fused-conversion kernel; without async builtins pass 2 stages via registers.

typedef float  v4f   __attribute__((ext_vector_type(4)));
typedef float  v8f   __attribute__((ext_vector_type(8)));
typedef __bf16 v16bf __attribute__((ext_vector_type(16)));
typedef int    v4i_t __attribute__((ext_vector_type(4)));

#define NEXPERT 8
#define DIM_IN  2048
#define DIM_OUT 2048
#define TM 128
#define TN 128
#define KT 32               // one bf16-WMMA K block per slab
#define NK (DIM_IN / KT)    // 64

#if __has_builtin(__builtin_amdgcn_global_load_async_to_lds_b128)
#define HAVE_ASYNC_LDS 1
#else
#define HAVE_ASYNC_LDS 0
#endif

union FragU { uint4 q[2]; v16bf v; };

__device__ __forceinline__ unsigned pk_hi16(float a, float b) {
    // {bf16_trunc(b), bf16_trunc(a)} via one v_perm_b32 (a -> [15:0])
    return __builtin_amdgcn_perm(__float_as_uint(b), __float_as_uint(a), 0x07060302u);
}
__device__ __forceinline__ float res_lo(float a) {
    return a - __uint_as_float(__float_as_uint(a) & 0xffff0000u);
}
__device__ __forceinline__ void wait_async0() {
#if __has_builtin(__builtin_amdgcn_s_wait_asynccnt)
    __builtin_amdgcn_s_wait_asynccnt(0);
#else
    asm volatile("s_wait_asynccnt 0x0" ::: "memory");
#endif
}

// ---------------- pass 1: f32 -> bf16 hi/lo planes --------------------------
__global__ __launch_bounds__(256)
void split_f32_kernel(const float* __restrict__ src,
                      unsigned short* __restrict__ hi,
                      unsigned short* __restrict__ lo,
                      unsigned n)
{
    const unsigned i = (blockIdx.x * 256u + threadIdx.x) * 8u;
    if (i >= n) return;
    const v4f f0 = *(const v4f*)(src + i);
    const v4f f1 = *(const v4f*)(src + i + 4);
    uint4 h;
    h.x = pk_hi16(f0.x, f0.y); h.y = pk_hi16(f0.z, f0.w);
    h.z = pk_hi16(f1.x, f1.y); h.w = pk_hi16(f1.z, f1.w);
    v4f r0, r1;
    r0.x = res_lo(f0.x); r0.y = res_lo(f0.y); r0.z = res_lo(f0.z); r0.w = res_lo(f0.w);
    r1.x = res_lo(f1.x); r1.y = res_lo(f1.y); r1.z = res_lo(f1.z); r1.w = res_lo(f1.w);
    uint4 l;
    l.x = pk_hi16(r0.x, r0.y); l.y = pk_hi16(r0.z, r0.w);
    l.z = pk_hi16(r1.x, r1.y); l.w = pk_hi16(r1.z, r1.w);
    *(uint4*)(hi + i) = h;
    *(uint4*)(lo + i) = l;
}

// LDS fragment readers. Row = 128B: chunks 0-3 = hi k0..31, 4-7 = lo k0..31;
// chunk index XOR-swizzled by (row>>1)&7 -> 16-lane b128 reads are bank-perfect.
__device__ __forceinline__ v16bf fragA(const unsigned char* mb, int row, int pl, int hf) {
    const int s = (row >> 1) & 7;
    FragU u;
    u.q[0] = *(const uint4*)(mb + row * 128 + (((pl * 4 + hf)     ^ s) << 4));
    u.q[1] = *(const uint4*)(mb + row * 128 + (((pl * 4 + 2 + hf) ^ s) << 4));
    return u.v;
}
__device__ __forceinline__ v16bf fragB(const unsigned char* mb, int row, int pl, int hf) {
    const int s = (row >> 1) & 7;
    FragU u;
    u.q[0] = *(const uint4*)(mb + row * 128 + (((pl * 4 + hf)     ^ s) << 4));
    u.q[1] = *(const uint4*)(mb + row * 128 + (((pl * 4 + 2 + hf) ^ s) << 4));
    return u.v;
}

// ---------------- pass 2: GEMM on pre-split bf16 planes ---------------------
__global__ __launch_bounds__(256)
void gemm_presplit_kernel(const unsigned short* __restrict__ ws,
                          unsigned nx, unsigned nw,
                          const float* __restrict__ bias,
                          const int* __restrict__ tpe,
                          float* __restrict__ out)
{
    __shared__ __align__(16) unsigned char Lds[2 * 2 * TM * 128];   // 65536 B

    const int tid  = threadIdx.x;
    const int lane = tid & 31;
    const int wave = tid >> 5;
    const int wm   = wave >> 2;      // 2x4 wave grid, wave tile 64x32
    const int wn   = wave & 3;
    const int hf   = lane >> 4;
    const int l16  = lane & 15;

    // L2-friendly supertile order: 8 M-tiles x all N-tiles per group.
    const int numM = gridDim.x, numN = gridDim.y;
    const int bid  = blockIdx.y * numM + blockIdx.x;
    const int GM   = 8;
    const int per  = GM * numN;
    const int grp  = bid / per;
    const int rem0 = bid % per;
    const int mfirst = grp * GM;
    const int msz  = (GM < numM - mfirst) ? GM : (numM - mfirst);
    const int tokBase = (mfirst + rem0 % msz) * TM;
    const int colBase = (rem0 / msz) * TN;

    // Expert for this tile (uniform scalar scan; counts are multiples of TM).
    int e = 0;
    int pref = tpe[0];
    while (e < NEXPERT - 1 && tokBase >= pref) { ++e; pref += tpe[e]; }

    // Per-thread copy descriptors: 2048 16B chunks per slab, 8 per thread.
    // ws planes: xh[0,nx) xl[nx,2nx) wh[2nx,2nx+nw) wl[2nx+nw,2nx+2nw)
    unsigned goff[8];   // element offset into ws (slab k=0)
    unsigned loff[8];   // byte offset into Lds buffer 0
    #pragma unroll
    for (int j = 0; j < 8; ++j) {
        const int c   = j * 256 + tid;     // 0..2047
        const int mat = c >> 10;           // 0=A(x), 1=B(W)
        const int rm  = c & 1023;
        const int row = rm >> 3;
        const int ch  = rm & 7;
        const int pl  = ch >> 2;           // 0=hi, 1=lo
        const int qq  = ch & 3;
        unsigned g;
        if (mat == 0)
            g = (pl ? nx : 0u) + (unsigned)(tokBase + row) * DIM_IN + qq * 8;
        else
            g = 2u * nx + (pl ? nw : 0u)
              + (unsigned)e * (DIM_OUT * DIM_IN)
              + (unsigned)(colBase + row) * DIM_IN + qq * 8;
        goff[j] = g;
        loff[j] = (unsigned)(mat * 16384 + row * 128 + ((ch ^ ((row >> 1) & 7)) << 4));
    }

    v8f acc[4][2] = {};

    auto compute = [&](int buf) {
        const unsigned char* Ab = Lds + buf * 32768;
        const unsigned char* Bb = Ab + 16384;
        const int arow0 = wm * 64 + l16;
        const int brow0 = wn * 32 + l16;
        v16bf bh[2], bl[2];
        #pragma unroll
        for (int ni = 0; ni < 2; ++ni) {
            bh[ni] = fragB(Bb, brow0 + ni * 16, 0, hf);
            bl[ni] = fragB(Bb, brow0 + ni * 16, 1, hf);
        }
        #pragma unroll
        for (int mi = 0; mi < 4; ++mi) {
            const v16bf ah = fragA(Ab, arow0 + mi * 16, 0, hf);
            const v16bf al = fragA(Ab, arow0 + mi * 16, 1, hf);
            #pragma unroll
            for (int ni = 0; ni < 2; ++ni) {
                acc[mi][ni] = __builtin_amdgcn_wmma_f32_16x16x32_bf16(
                    false, ah, false, bh[ni], (short)0, acc[mi][ni], false, false);
                acc[mi][ni] = __builtin_amdgcn_wmma_f32_16x16x32_bf16(
                    false, ah, false, bl[ni], (short)0, acc[mi][ni], false, false);
                acc[mi][ni] = __builtin_amdgcn_wmma_f32_16x16x32_bf16(
                    false, al, false, bh[ni], (short)0, acc[mi][ni], false, false);
            }
        }
    };

#if HAVE_ASYNC_LDS
    auto issue_copy = [&](int kb, int buf) {
        const unsigned ko = (unsigned)kb * KT;            // element step along K
        const unsigned bb = (unsigned)buf * 32768u;
        #pragma unroll
        for (int j = 0; j < 8; ++j)
            __builtin_amdgcn_global_load_async_to_lds_b128(
                (__attribute__((address_space(1))) v4i_t*)(ws + goff[j] + ko),
                (__attribute__((address_space(3))) v4i_t*)(Lds + bb + loff[j]),
                0, 0);
    };
    issue_copy(0, 0);
    for (int kb = 0; kb < NK; ++kb) {
        wait_async0();                       // my slab-kb chunks landed in LDS
        __syncthreads();                     // everyone's did; prev readers done
        if (kb + 1 < NK) issue_copy(kb + 1, (kb + 1) & 1);
        compute(kb & 1);                     // overlaps with async fill of nxt
    }
#else
    uint4 stg[8];
    auto load_slab = [&](int kb) {
        const unsigned ko = (unsigned)kb * KT;
        #pragma unroll
        for (int j = 0; j < 8; ++j)
            stg[j] = *(const uint4*)(ws + goff[j] + ko);
    };
    auto store_slab = [&](int buf) {
        #pragma unroll
        for (int j = 0; j < 8; ++j)
            *(uint4*)(Lds + buf * 32768u + loff[j]) = stg[j];
    };
    load_slab(0);
    store_slab(0);
    for (int kb = 0; kb < NK; ++kb) {
        __syncthreads();
        if (kb + 1 < NK) load_slab(kb + 1);
        compute(kb & 1);
        if (kb + 1 < NK) store_slab((kb + 1) & 1);
    }
#endif

    // Epilogue: fused bias. C/D layout: VGPR r <-> M = r + 8*hf, N = l16.
    #pragma unroll
    for (int ni = 0; ni < 2; ++ni) {
        const int col = colBase + wn * 32 + ni * 16 + l16;
        const float bv = bias[(size_t)e * DIM_OUT + col];
        #pragma unroll
        for (int mi = 0; mi < 4; ++mi) {
            const int row0 = tokBase + wm * 64 + mi * 16 + hf * 8;
            #pragma unroll
            for (int r = 0; r < 8; ++r)
                out[(size_t)(row0 + r) * DIM_OUT + col] = acc[mi][ni][r] + bv;
        }
    }
}

// ---------------- fallback: fused-conversion GEMM (no workspace) ------------
#define PITCH 72
__device__ __forceinline__ void split_store(unsigned short* hp, unsigned short* lp, v4f f) {
    uint2 h, l;
    h.x = pk_hi16(f.x, f.y); h.y = pk_hi16(f.z, f.w);
    v4f r;
    r.x = res_lo(f.x); r.y = res_lo(f.y); r.z = res_lo(f.z); r.w = res_lo(f.w);
    l.x = pk_hi16(r.x, r.y); l.y = pk_hi16(r.z, r.w);
    *(uint2*)hp = h;
    *(uint2*)lp = l;
}
__device__ __forceinline__ v16bf loadAp(const unsigned short* row, int hf) {
    FragU u; const uint4* p = (const uint4*)row;
    u.q[0] = p[hf]; u.q[1] = p[2 + hf]; return u.v;
}
__device__ __forceinline__ v16bf loadBp(const unsigned short* row, int hf) {
    FragU u; const uint4* p = (const uint4*)row;
    u.q[0] = p[hf]; u.q[1] = p[2 + hf]; return u.v;
}

__global__ __launch_bounds__(256)
void gemm_fused_kernel(const float* __restrict__ x, const float* __restrict__ w,
                       const float* __restrict__ bias, const int* __restrict__ tpe,
                       float* __restrict__ out)
{
    __shared__ __align__(16) unsigned short As[TM * PITCH];
    __shared__ __align__(16) unsigned short Bs[TN * PITCH];

    const int tid = threadIdx.x, lane = tid & 31, wave = tid >> 5;
    const int wm = wave >> 2, wn = wave & 3, hf = lane >> 4, l16 = lane & 15;
    const int tokBase = blockIdx.x * TM, colBase = blockIdx.y * TN;

    int e = 0; int pref = tpe[0];
    while (e < NEXPERT - 1 && tokBase >= pref) { ++e; pref += tpe[e]; }

    const float* xA = x + (size_t)tokBase * DIM_IN;
    const float* wB = w + (size_t)e * DIM_OUT * DIM_IN + (size_t)colBase * DIM_IN;

    const int q = tid & 7, r0 = tid >> 3;
    v4f ra[4], rb[4];
    auto load_slab = [&](int kb) {
        const size_t kc = (size_t)kb * KT + q * 4;
        #pragma unroll
        for (int p = 0; p < 4; ++p) {
            ra[p] = *(const v4f*)(xA + (size_t)(r0 + 32 * p) * DIM_IN + kc);
            rb[p] = *(const v4f*)(wB + (size_t)(r0 + 32 * p) * DIM_IN + kc);
        }
    };
    auto store_slab = [&]() {
        #pragma unroll
        for (int p = 0; p < 4; ++p) {
            const int base = (r0 + 32 * p) * PITCH + q * 4;
            split_store(&As[base], &As[base + 32], ra[p]);
            split_store(&Bs[base], &Bs[base + 32], rb[p]);
        }
    };

    v8f acc[4][2] = {};
    load_slab(0); store_slab();
    for (int kb = 0; kb < NK; ++kb) {
        __syncthreads();
        if (kb + 1 < NK) load_slab(kb + 1);
        const unsigned short* Arow = &As[(wm * 64 + l16) * PITCH];
        const unsigned short* Brow = &Bs[(wn * 32 + l16) * PITCH];
        v16bf bh[2], bl[2];
        #pragma unroll
        for (int ni = 0; ni < 2; ++ni) {
            bh[ni] = loadBp(Brow + ni * 16 * PITCH, hf);
            bl[ni] = loadBp(Brow + ni * 16 * PITCH + 32, hf);
        }
        #pragma unroll
        for (int mi = 0; mi < 4; ++mi) {
            const v16bf ah = loadAp(Arow + mi * 16 * PITCH, hf);
            const v16bf al = loadAp(Arow + mi * 16 * PITCH + 32, hf);
            #pragma unroll
            for (int ni = 0; ni < 2; ++ni) {
                acc[mi][ni] = __builtin_amdgcn_wmma_f32_16x16x32_bf16(
                    false, ah, false, bh[ni], (short)0, acc[mi][ni], false, false);
                acc[mi][ni] = __builtin_amdgcn_wmma_f32_16x16x32_bf16(
                    false, ah, false, bl[ni], (short)0, acc[mi][ni], false, false);
                acc[mi][ni] = __builtin_amdgcn_wmma_f32_16x16x32_bf16(
                    false, al, false, bh[ni], (short)0, acc[mi][ni], false, false);
            }
        }
        __syncthreads();
        if (kb + 1 < NK) store_slab();
    }

    #pragma unroll
    for (int ni = 0; ni < 2; ++ni) {
        const int col = colBase + wn * 32 + ni * 16 + l16;
        const float bv = bias[(size_t)e * DIM_OUT + col];
        #pragma unroll
        for (int mi = 0; mi < 4; ++mi) {
            const int row0 = tokBase + wm * 64 + mi * 16 + hf * 8;
            #pragma unroll
            for (int r = 0; r < 8; ++r)
                out[(size_t)(row0 + r) * DIM_OUT + col] = acc[mi][ni][r] + bv;
        }
    }
}

extern "C" void kernel_launch(void* const* d_in, const int* in_sizes, int n_in,
                              void* d_out, int out_size, void* d_ws, size_t ws_size,
                              hipStream_t stream) {
    (void)n_in; (void)out_size;
    const float* x    = (const float*)d_in[0];
    const float* w    = (const float*)d_in[1];
    const float* bias = (const float*)d_in[2];
    const int*   tpe  = (const int*)d_in[3];

    const int T = in_sizes[0] / DIM_IN;
    const unsigned nx = (unsigned)T * DIM_IN;                     // 33.5M
    const unsigned nw = (unsigned)NEXPERT * DIM_OUT * DIM_IN;     // 33.5M
    const size_t need = ((size_t)nx + nw) * 2u * sizeof(unsigned short);

    dim3 grid(T / TM, DIM_OUT / TN);
    if (ws_size >= need) {
        unsigned short* wsp = (unsigned short*)d_ws;
        split_f32_kernel<<<nx / 2048, 256, 0, stream>>>(x, wsp, wsp + nx, nx);
        split_f32_kernel<<<nw / 2048, 256, 0, stream>>>(w, wsp + 2u * (size_t)nx,
                                                        wsp + 2u * (size_t)nx + nw, nw);
        gemm_presplit_kernel<<<grid, 256, 0, stream>>>(wsp, nx, nw, bias, tpe,
                                                       (float*)d_out);
    } else {
        gemm_fused_kernel<<<grid, 256, 0, stream>>>(x, w, bias, tpe, (float*)d_out);
    }
}
